// LConv2d_72249939853688
// MI455X (gfx1250) — matrix operand order repaired and checked by reading, not compiled, not run on hardware
//
#include <hip/hip_runtime.h>
#include <hip/hip_bf16.h>

#define BATCH 8
#define CIN   128
#define COUT  128
#define LSZ   128
#define TS    16          // sites per tile (along contiguous x)
#define CPAD  136         // padded channel pitch in LDS (u16 units) -> 272B row

typedef __attribute__((ext_vector_type(16))) __bf16         v16bf;
typedef __attribute__((ext_vector_type(16))) unsigned short v16u;
typedef __attribute__((ext_vector_type(8)))  unsigned short v8u;
typedef __attribute__((ext_vector_type(8)))  float          v8f;

// round-to-nearest-even bf16 sits in the HIGH half of the returned u32
__device__ __forceinline__ unsigned int rnd_bf16_hi(float f) {
    unsigned int u = __float_as_uint(f);
    return u + 0x7FFFu + ((u >> 16) & 1u);
}
// pack bf16(lo) into [15:0], bf16(hi) into [31:16] with one v_perm_b32
__device__ __forceinline__ unsigned int pack_bf16(float lo, float hi) {
    return __builtin_amdgcn_perm(rnd_bf16_hi(hi), rnd_bf16_hi(lo), 0x07060302u);
}

// ---------------------------------------------------------------------------
// One-shot prep: convert 5 weight matrices (f32) into bf16, pre-swizzled into
// the exact per-lane WMMA A-operand layout:
//   slot id = ((kk*5 + n)*8 + wave)*32 + lane, 32 bytes per slot.
//   lane<16 : K {cb..cb+7, cb+16..cb+23}, lane>=16: K {cb+8..cb+15, cb+24..31}
// ---------------------------------------------------------------------------
__global__ __launch_bounds__(256)
void prep_weights(const float* __restrict__ wc,
                  const float* __restrict__ wf,
                  const float* __restrict__ wb,
                  unsigned short* __restrict__ wsw)
{
    const int id   = blockIdx.x * 256 + threadIdx.x;   // 0..5119
    const int lane = id & 31;
    const int wv   = (id >> 5) & 7;
    const int t    = id >> 8;          // kk*5 + n
    const int n    = t % 5;
    const int kk   = t / 5;

    const float* wsel[5] = { wc, wf, wb, wf + COUT * CIN, wb + COUT * CIN };

    const int lhalf = lane >> 4;
    const int lmod  = lane & 15;
    const int orow  = wv * 16 + lmod;
    const int cbA   = kk * 32 + lhalf * 8;

    const float* Wr = wsel[n] + (size_t)orow * CIN;
    const float4 w0 = *(const float4*)(Wr + cbA);
    const float4 w1 = *(const float4*)(Wr + cbA + 4);
    const float4 w2 = *(const float4*)(Wr + cbA + 16);
    const float4 w3 = *(const float4*)(Wr + cbA + 20);

    uint4 q0, q1;
    q0.x = pack_bf16(w0.x, w0.y);  q0.y = pack_bf16(w0.z, w0.w);
    q0.z = pack_bf16(w1.x, w1.y);  q0.w = pack_bf16(w1.z, w1.w);
    q1.x = pack_bf16(w2.x, w2.y);  q1.y = pack_bf16(w2.z, w2.w);
    q1.z = pack_bf16(w3.x, w3.y);  q1.w = pack_bf16(w3.z, w3.w);

    uint4* dst = (uint4*)(wsw + (size_t)id * 16);
    dst[0] = q0;
    dst[1] = q1;
}

// ---------------------------------------------------------------------------
// Main kernel
// ---------------------------------------------------------------------------
__global__ __launch_bounds__(256)
void lconv2d_wmma(const float* __restrict__ fr,
                  const float* __restrict__ fi,
                  const float* __restrict__ links,
                  const unsigned short* __restrict__ wsw,
                  float* __restrict__ outr,
                  float* __restrict__ outi)
{
    // 10 tiles: [stencil n (5)][comp r/i (2)] x [site 16][channel 128 (+pad)]
    __shared__ __align__(64) unsigned short sF[10][TS][CPAD];

    const int tid = threadIdx.x;
    const int x0  = blockIdx.x * TS;
    const int y0  = blockIdx.y;
    const int b   = blockIdx.z;

    // ---------------- Phase 1: stage bf16 feature tiles into LDS ----------
    // thread -> (site s, channel-group cg of 8); loop over the 10 tiles.
    {
        const int s  = tid & 15;
        const int cg = tid >> 4;            // 0..15 -> channels cg*8 .. cg*8+7
        #pragma unroll
        for (int d = 0; d < 10; ++d) {
            const int n  = d >> 1;          // 0=c, 1=+y, 2=-y, 3=+x, 4=-x
            const int dy = (n == 1) - (n == 2);
            const int dx = (n == 3) - (n == 4);
            const float* f = (d & 1) ? fi : fr;
            const int y = (y0 + dy) & (LSZ - 1);
            const int x = (x0 + s + dx) & (LSZ - 1);
            const float* p = f + (((size_t)(b * CIN + cg * 8)) << 14) + (y << 7) + x;
            float v0 = p[0 << 14], v1 = p[1 << 14], v2 = p[2 << 14], v3 = p[3 << 14];
            float v4 = p[4 << 14], v5 = p[5 << 14], v6 = p[6 << 14], v7 = p[7 << 14];
            uint4 q;
            q.x = pack_bf16(v0, v1);
            q.y = pack_bf16(v2, v3);
            q.z = pack_bf16(v4, v5);
            q.w = pack_bf16(v6, v7);
            *(uint4*)&sF[d][s][cg * 8] = q;
        }
    }
    __syncthreads();

    // ---------------- Phase 2: WMMA GEMMs ----------------------------------
    const int wave  = tid >> 5;
    const int lane  = tid & 31;
    const int lhalf = lane >> 4;
    const int lmod  = lane & 15;

    const v8f zero = {0.f, 0.f, 0.f, 0.f, 0.f, 0.f, 0.f, 0.f};
    v8f acc[10];
    #pragma unroll
    for (int i = 0; i < 10; ++i) acc[i] = zero;

    // pre-swizzled A operands: one 32B load per (kk, n)
    const unsigned short* wbase = wsw + (size_t)(wave * 32 + lane) * 16;

    #pragma unroll
    for (int kk = 0; kk < 4; ++kk) {
        const int cbB = kk * 32 + lhalf * 8;
        #pragma unroll
        for (int n = 0; n < 5; ++n) {
            const v16u au = *(const v16u*)(wbase + (kk * 5 + n) * 4096);
            const v16bf a = __builtin_bit_cast(v16bf, au);

            const v8u brlo = *(const v8u*)&sF[n * 2 + 0][lmod][cbB];
            const v8u brhi = *(const v8u*)&sF[n * 2 + 0][lmod][cbB + 16];
            const v8u bilo = *(const v8u*)&sF[n * 2 + 1][lmod][cbB];
            const v8u bihi = *(const v8u*)&sF[n * 2 + 1][lmod][cbB + 16];
            const v16bf br = __builtin_bit_cast(v16bf,
                __builtin_shufflevector(brlo, brhi, 0,1,2,3,4,5,6,7,8,9,10,11,12,13,14,15));
            const v16bf bi = __builtin_bit_cast(v16bf,
                __builtin_shufflevector(bilo, bihi, 0,1,2,3,4,5,6,7,8,9,10,11,12,13,14,15));

            acc[n * 2 + 0] = __builtin_amdgcn_wmma_f32_16x16x32_bf16(
                false, a, false, br, (short)0, acc[n * 2 + 0], false, false);
            acc[n * 2 + 1] = __builtin_amdgcn_wmma_f32_16x16x32_bf16(
                false, a, false, bi, (short)0, acc[n * 2 + 1], false, false);
        }
    }

    // ---------------- Phase 3: link-phase epilogue + store -----------------
    const int xs = x0 + lmod;
    const int xm = (xs - 1) & (LSZ - 1);
    const int ym = (y0 - 1) & (LSZ - 1);
    const size_t lb = (size_t)(b * 4) << 14;

    const float lr0 =  links[lb + (0 << 14) + (y0 << 7) + xs];
    const float li0 =  links[lb + (1 << 14) + (y0 << 7) + xs];
    const float lr1 =  links[lb + (2 << 14) + (y0 << 7) + xs];
    const float li1 =  links[lb + (3 << 14) + (y0 << 7) + xs];
    const float cr0 =  links[lb + (0 << 14) + (ym << 7) + xs];   // conj(U) rolled +1 in y
    const float ci0 = -links[lb + (1 << 14) + (ym << 7) + xs];
    const float cr1 =  links[lb + (2 << 14) + (y0 << 7) + xm];   // conj(U) rolled +1 in x
    const float ci1 = -links[lb + (3 << 14) + (y0 << 7) + xm];

    #pragma unroll
    for (int j = 0; j < 8; ++j) {
        const int o = wave * 16 + j + lhalf * 8;   // C/D layout: lanes16-31 -> M=j+8
        const float gcr = acc[0][j], gci = acc[1][j];
        const float f0r = acc[2][j], f0i = acc[3][j];
        const float b0r = acc[4][j], b0i = acc[5][j];
        const float f1r = acc[6][j], f1i = acc[7][j];
        const float b1r = acc[8][j], b1i = acc[9][j];

        const float orv = gcr + lr0 * f0r - li0 * f0i + cr0 * b0r - ci0 * b0i
                              + lr1 * f1r - li1 * f1i + cr1 * b1r - ci1 * b1i;
        const float oiv = gci + lr0 * f0i + li0 * f0r + cr0 * b0i + ci0 * b0r
                              + lr1 * f1i + li1 * f1r + cr1 * b1i + ci1 * b1r;

        const size_t oidx = ((size_t)(b * COUT + o) << 14) + (y0 << 7) + xs;
        outr[oidx] = orv;
        outi[oidx] = oiv;
    }
}

extern "C" void kernel_launch(void* const* d_in, const int* in_sizes, int n_in,
                              void* d_out, int out_size, void* d_ws, size_t ws_size,
                              hipStream_t stream) {
    const float* fr    = (const float*)d_in[0];
    const float* fi    = (const float*)d_in[1];
    const float* links = (const float*)d_in[2];
    const float* wc    = (const float*)d_in[3];
    const float* wf    = (const float*)d_in[4];
    const float* wb    = (const float*)d_in[5];
    float* outr = (float*)d_out;
    float* outi = outr + (size_t)BATCH * COUT * LSZ * LSZ;
    unsigned short* wsw = (unsigned short*)d_ws;   // 160 KB of pre-swizzled bf16 weights

    // one-shot weight swizzle+convert (5120 lane-slots)
    prep_weights<<<20, 256, 0, stream>>>(wc, wf, wb, wsw);

    dim3 grid(LSZ / TS, LSZ, BATCH);
    lconv2d_wmma<<<grid, 256, 0, stream>>>(fr, fi, links, wsw, outr, outi);
}
